// GroupedQueryAttention_87325275062890
// MI455X (gfx1250) — matrix operand order repaired and checked by reading, not compiled, not run on hardware
//
#include <hip/hip_runtime.h>
#include <hip/hip_bf16.h>
#include <math.h>

// ---------------------------------------------------------------------------
// GroupedQueryAttention for MI455X (gfx1250, wave32, WMMA)
// f16 data path with f32 WMMA accumulation (v_wmma_f32_16x16x32_f16)
// K/V tiles staged with GLOBAL_LOAD_ASYNC_TO_LDS_B128 (ASYNCcnt path)
// ---------------------------------------------------------------------------

typedef _Float16 half_t;
typedef __attribute__((ext_vector_type(16))) _Float16 v16h;
typedef __attribute__((ext_vector_type(8)))  float    v8f;

#define S_LEN 2048
#define DIM_  2048
#define HD    128
#define QH    16
#define KVH_  4
#define KVD   512

static __device__ __forceinline__ v8f wmma16(v16h a, v16h b, v8f c) {
  // (neg_a, A, neg_b, B, c_mod, C, reuse_a, reuse_b)
  return __builtin_amdgcn_wmma_f32_16x16x32_f16(false, a, false, b, (short)0, c,
                                                false, false);
}

// A-fragment (16x32 f16, wave32): lane L holds row m=L&15; VGPR v pair base:
//   kbase(v) = (L>>4)*16 + v*2
static __device__ __forceinline__ int a_kbase(int v, int hlf) {
  return (hlf << 4) + (v << 1);
}

// ---------------------------------------------------------------------------
// GEMM: C[M][N] = A[M][K] @ W[N][K]^T + bias[N]   (A fp32 or f16, W fp32)
// block tile 128x128, 8 waves as 4(M) x 2(N), wave tile 32x64
// ---------------------------------------------------------------------------
template <bool A_HALF>
__global__ __launch_bounds__(256) void gemm_bias_kernel(
    const void* __restrict__ Ap, const float* __restrict__ W,
    const float* __restrict__ bias, float* __restrict__ C, int M, int N,
    int K) {
  __shared__ __align__(16) half_t As[128 * 32];
  __shared__ __align__(16) half_t Ws[128 * 32];

  const int tid = threadIdx.x;
  const int lane = tid & 31;
  const int wave = tid >> 5;
  const int hlf = lane >> 4;
  const int ln = lane & 15;
  const int wm = wave >> 1;  // 0..3 -> M offset wm*32
  const int wn = wave & 1;   // 0..1 -> N offset wn*64
  const int bm = blockIdx.y * 128;
  const int bn = blockIdx.x * 128;

  const float* Af = (const float*)Ap;
  const half_t* Ah = (const half_t*)Ap;

  v8f acc[2][4];
  v8f zero = {0.f, 0.f, 0.f, 0.f, 0.f, 0.f, 0.f, 0.f};
#pragma unroll
  for (int i = 0; i < 2; i++)
#pragma unroll
    for (int j = 0; j < 4; j++) acc[i][j] = zero;

  const int stg_r = (tid * 16) >> 5;   // row within tile for this thread
  const int stg_c = (tid * 16) & 31;   // 16-elem aligned column base

  for (int kt = 0; kt < K; kt += 32) {
    __syncthreads();
    // stage A tile [128 x 32] and W tile [128 x 32] as f16
    {
      const size_t abase = (size_t)(bm + stg_r) * K + kt + stg_c;
      const size_t wbase = (size_t)(bn + stg_r) * K + kt + stg_c;
      const int sbase = stg_r * 32 + stg_c;
#pragma unroll
      for (int e = 0; e < 16; e++) {
        As[sbase + e] = A_HALF ? Ah[abase + e] : (half_t)Af[abase + e];
        Ws[sbase + e] = (half_t)W[wbase + e];
      }
      if (kt + 32 < K) {
        if (A_HALF)
          __builtin_prefetch(&Ah[abase + 32], 0, 1);
        else
          __builtin_prefetch(&Af[abase + 32], 0, 1);
        __builtin_prefetch(&W[wbase + 32], 0, 1);
      }
    }
    __syncthreads();

    // hoist all fragments, then issue the 8 WMMAs back-to-back
    v16h a[2], b[4];
#pragma unroll
    for (int mi = 0; mi < 2; mi++) {
      const int row = wm * 32 + mi * 16 + ln;
#pragma unroll
      for (int v = 0; v < 8; v++) {
        const int kb = a_kbase(v, hlf);
        a[mi][2 * v] = As[row * 32 + kb];
        a[mi][2 * v + 1] = As[row * 32 + kb + 1];
      }
    }
    const int kof = hlf << 4;  // B frag: lanes 0-15 K=2v(+1); lanes 16-31 K=16+2v(+1)
#pragma unroll
    for (int ni = 0; ni < 4; ni++) {
      const int nrow = wn * 64 + ni * 16 + ln;
#pragma unroll
      for (int v = 0; v < 8; v++) {
        b[ni][2 * v] = Ws[nrow * 32 + kof + 2 * v];
        b[ni][2 * v + 1] = Ws[nrow * 32 + kof + 2 * v + 1];
      }
    }
#pragma unroll
    for (int mi = 0; mi < 2; mi++)
#pragma unroll
      for (int ni = 0; ni < 4; ni++) acc[mi][ni] = wmma16(a[mi], b[ni], acc[mi][ni]);
  }

  // epilogue: C/D layout row = v + 8*hlf, col = ln
#pragma unroll
  for (int mi = 0; mi < 2; mi++)
#pragma unroll
    for (int ni = 0; ni < 4; ni++) {
      const int col = bn + wn * 64 + ni * 16 + ln;
      const float bb = bias[col];
#pragma unroll
      for (int v = 0; v < 8; v++) {
        const int row = bm + wm * 32 + mi * 16 + v + (hlf << 3);
        C[(size_t)row * N + col] = acc[mi][ni][v] + bb;
      }
    }
}

// ---------------------------------------------------------------------------
// RoPE + pack to head-major f16: q16[QH][S][HD], k16[KVH][S][HD], v16[KVH][S][HD]
// ---------------------------------------------------------------------------
__global__ __launch_bounds__(256) void rope_pack_kernel(
    const float* __restrict__ qr, const float* __restrict__ kr,
    const float* __restrict__ vr, half_t* __restrict__ q16,
    half_t* __restrict__ k16, half_t* __restrict__ v16) {
  const int s = blockIdx.x;
  const int tid = threadIdx.x;

  // adjusted base (reference: seq_eff = max(S, MAX_LEN) = 2048)
  const float t = 8.0f * (float)S_LEN / 2048.0f - 7.0f;
  const float ab = 10000.0f * powf(t, (float)HD / (float)(HD - 2));

  for (int idx = tid; idx < DIM_; idx += 256) {
    const int h = idx >> 7;
    const int d = idx & 127;
    const int d2 = d & 63;
    const float freq = (float)s * powf(ab, -(float)d2 / 64.0f);
    const float cs = cosf(freq), sn = sinf(freq);
    const float x = qr[(size_t)s * DIM_ + idx];
    const float pr = (d < 64) ? -qr[(size_t)s * DIM_ + h * HD + d + 64]
                              : qr[(size_t)s * DIM_ + h * HD + d - 64];
    q16[((size_t)h * S_LEN + s) * HD + d] = (half_t)(x * cs + pr * sn);
  }
  for (int idx = tid; idx < KVD; idx += 256) {
    const int h = idx >> 7;
    const int d = idx & 127;
    const int d2 = d & 63;
    const float freq = (float)s * powf(ab, -(float)d2 / 64.0f);
    const float cs = cosf(freq), sn = sinf(freq);
    const float x = kr[(size_t)s * KVD + idx];
    const float pr = (d < 64) ? -kr[(size_t)s * KVD + h * HD + d + 64]
                              : kr[(size_t)s * KVD + h * HD + d - 64];
    k16[((size_t)h * S_LEN + s) * HD + d] = (half_t)(x * cs + pr * sn);
  }
  for (int idx = tid; idx < KVD; idx += 256) {
    v16[((size_t)(idx >> 7) * S_LEN + s) * HD + (idx & 127)] =
        (half_t)vr[(size_t)s * KVD + idx];
  }
}

// ---------------------------------------------------------------------------
// Fused causal flash attention (GQA, rep=4): one block per (q-block-128, head)
// 8 waves x 16 query rows; 32-key K/V tiles staged by async LDS DMA.
// Writes att16[S][DIM] = softmax(QK^T/sqrt(d)) V  in f16, head-major columns.
// ---------------------------------------------------------------------------
__global__ __launch_bounds__(256) void attention_kernel(
    const half_t* __restrict__ q16, const half_t* __restrict__ k16,
    const half_t* __restrict__ v16, half_t* __restrict__ att16) {
  __shared__ __align__(16) half_t Kt[32 * HD];
  __shared__ __align__(16) half_t Vt[32 * HD];
  __shared__ __align__(16) half_t Pb[8][16][32];

  const int qb = blockIdx.x;   // 0..15, 128 query rows each
  const int h = blockIdx.y;    // 0..15
  const int kvh = h >> 2;      // GQA: 4 q-heads share one kv-head
  const int tid = threadIdx.x;
  const int lane = tid & 31;
  const int wave = tid >> 5;
  const int hlf = lane >> 4;
  const int ln = lane & 15;
  const int r0 = qb * 128 + wave * 16;
  const float scale = 0.08838834764831845f;  // 1/sqrt(128)

  // LDS byte offsets for this thread's async-DMA destinations (16B per lane)
  const unsigned lkt = (unsigned)(size_t)Kt + (unsigned)(tid * 16);
  const unsigned lvt = (unsigned)(size_t)Vt + (unsigned)(tid * 16);

  // Q fragments for the wave's 16 rows: 4 K-steps of 32 across HD=128
  v16h aq[4];
  {
    const half_t* qrow = q16 + ((size_t)h * S_LEN + (r0 + ln)) * HD;
#pragma unroll
    for (int kk = 0; kk < 4; kk++)
#pragma unroll
      for (int v = 0; v < 8; v++) {
        const int kb = a_kbase(v, hlf);
        aq[kk][2 * v] = qrow[kk * 32 + kb];
        aq[kk][2 * v + 1] = qrow[kk * 32 + kb + 1];
      }
  }

  v8f zero = {0.f, 0.f, 0.f, 0.f, 0.f, 0.f, 0.f, 0.f};
  v8f o[8];
  float mrow[8], lrow[8], al8[8];
#pragma unroll
  for (int v = 0; v < 8; v++) {
    o[v] = zero;
    mrow[v] = -1.0e30f;
    lrow[v] = 0.0f;
  }

  const int nkb = (qb + 1) * 4;  // 32-key blocks up to the causal diagonal
  for (int kb = 0; kb < nkb; kb++) {
    const int j0 = kb * 32;
    __syncthreads();  // previous iteration's LDS reads complete

    // Async DMA: 8KB K tile + 8KB V tile, 2 x b128 per thread per tile.
    // INST_OFFSET applies to both the LDS and global addresses (ISA 10.7).
    {
      const unsigned long long gka =
          (unsigned long long)(size_t)(k16 + ((size_t)kvh * S_LEN + j0) * HD) +
          (unsigned long long)(tid * 16);
      const unsigned long long gva =
          (unsigned long long)(size_t)(v16 + ((size_t)kvh * S_LEN + j0) * HD) +
          (unsigned long long)(tid * 16);
      asm volatile(
          "global_load_async_to_lds_b128 %0, %2, off\n\t"
          "global_load_async_to_lds_b128 %0, %2, off offset:4096\n\t"
          "global_load_async_to_lds_b128 %1, %3, off\n\t"
          "global_load_async_to_lds_b128 %1, %3, off offset:4096\n\t"
          "s_wait_asynccnt 0"
          :: "v"(lkt), "v"(lvt), "v"(gka), "v"(gva)
          : "memory");
    }
    __syncthreads();

    // scores S = Q K^T : hoist all 8 B frags, then 8 WMMAs back-to-back
    const int kof = hlf << 4;
    v16h bk[2][4];
#pragma unroll
    for (int nt = 0; nt < 2; nt++)
#pragma unroll
      for (int kk = 0; kk < 4; kk++)
#pragma unroll
        for (int v = 0; v < 8; v++) {
          const int dd = kk * 32 + kof + 2 * v;
          bk[nt][kk][2 * v] = Kt[(nt * 16 + ln) * HD + dd];
          bk[nt][kk][2 * v + 1] = Kt[(nt * 16 + ln) * HD + dd + 1];
        }
    v8f c0 = zero, c1 = zero;
#pragma unroll
    for (int kk = 0; kk < 4; kk++) {
      c0 = wmma16(aq[kk], bk[0][kk], c0);
      c1 = wmma16(aq[kk], bk[1][kk], c1);
    }

    // causal mask + online softmax (row r = v + 8*hlf lives across 16 lanes)
#pragma unroll
    for (int v = 0; v < 8; v++) {
      const int i = r0 + v + (hlf << 3);
      float s0 = c0[v] * scale;
      float s1 = c1[v] * scale;
      if (j0 + ln > i) s0 = -1.0e30f;
      if (j0 + 16 + ln > i) s1 = -1.0e30f;
      float sm = fmaxf(s0, s1);
#pragma unroll
      for (int off = 1; off < 16; off <<= 1) sm = fmaxf(sm, __shfl_xor(sm, off, 16));
      const float mn = fmaxf(mrow[v], sm);
      const float al = __expf(mrow[v] - mn);
      const float p0 = __expf(s0 - mn);
      const float p1 = __expf(s1 - mn);
      float rs = p0 + p1;
#pragma unroll
      for (int off = 1; off < 16; off <<= 1) rs += __shfl_xor(rs, off, 16);
      lrow[v] = lrow[v] * al + rs;
      mrow[v] = mn;
      al8[v] = al;
      Pb[wave][v + (hlf << 3)][ln] = (half_t)p0;
      Pb[wave][v + (hlf << 3)][16 + ln] = (half_t)p1;
    }
#pragma unroll
    for (int dt = 0; dt < 8; dt++)
#pragma unroll
      for (int v = 0; v < 8; v++) o[dt][v] *= al8[v];

    __syncthreads();  // publish P (LDS) across lanes of each wave

    // O += P @ V : reload P in A-fragment layout; hoist V frags, then 8 WMMAs
    v16h pa;
#pragma unroll
    for (int v = 0; v < 8; v++) {
      const int kb2 = a_kbase(v, hlf);
      pa[2 * v] = Pb[wave][ln][kb2];
      pa[2 * v + 1] = Pb[wave][ln][kb2 + 1];
    }
    v16h bvf[8];
#pragma unroll
    for (int dt = 0; dt < 8; dt++)
#pragma unroll
      for (int v = 0; v < 8; v++) {
        bvf[dt][2 * v] = Vt[(kof + 2 * v) * HD + dt * 16 + ln];
        bvf[dt][2 * v + 1] = Vt[(kof + 2 * v + 1) * HD + dt * 16 + ln];
      }
#pragma unroll
    for (int dt = 0; dt < 8; dt++) o[dt] = wmma16(pa, bvf[dt], o[dt]);
  }

  // normalize and write out (s-major, head-major columns)
#pragma unroll
  for (int v = 0; v < 8; v++) {
    const float inv = 1.0f / lrow[v];
    const int i = r0 + v + (hlf << 3);
#pragma unroll
    for (int dt = 0; dt < 8; dt++) {
      att16[(size_t)i * DIM_ + h * HD + dt * 16 + ln] = (half_t)(o[dt][v] * inv);
    }
  }
}

// ---------------------------------------------------------------------------
extern "C" void kernel_launch(void* const* d_in, const int* in_sizes, int n_in,
                              void* d_out, int out_size, void* d_ws,
                              size_t ws_size, hipStream_t stream) {
  (void)in_sizes; (void)n_in; (void)out_size; (void)ws_size;
  const float* query = (const float*)d_in[0];
  const float* key = (const float*)d_in[1];
  const float* value = (const float*)d_in[2];
  const float* Wq = (const float*)d_in[3];
  const float* bq = (const float*)d_in[4];
  const float* Wk = (const float*)d_in[5];
  const float* bk = (const float*)d_in[6];
  const float* Wv = (const float*)d_in[7];
  const float* bv = (const float*)d_in[8];
  const float* Wo = (const float*)d_in[9];
  const float* bo = (const float*)d_in[10];
  float* out = (float*)d_out;

  char* ws = (char*)d_ws;
  float* q_raw = (float*)ws;  ws += sizeof(float) * (size_t)S_LEN * DIM_;
  float* k_raw = (float*)ws;  ws += sizeof(float) * (size_t)S_LEN * KVD;
  float* v_raw = (float*)ws;  ws += sizeof(float) * (size_t)S_LEN * KVD;
  half_t* q16 = (half_t*)ws;  ws += sizeof(half_t) * (size_t)QH * S_LEN * HD;
  half_t* k16 = (half_t*)ws;  ws += sizeof(half_t) * (size_t)KVH_ * S_LEN * HD;
  half_t* v16 = (half_t*)ws;  ws += sizeof(half_t) * (size_t)KVH_ * S_LEN * HD;
  half_t* att16 = (half_t*)ws;

  dim3 blk(256);
  // projections (fp32 in -> f16 staging -> f32 out)
  gemm_bias_kernel<false><<<dim3(DIM_ / 128, S_LEN / 128), blk, 0, stream>>>(
      query, Wq, bq, q_raw, S_LEN, DIM_, DIM_);
  gemm_bias_kernel<false><<<dim3(KVD / 128, S_LEN / 128), blk, 0, stream>>>(
      key, Wk, bk, k_raw, S_LEN, KVD, DIM_);
  gemm_bias_kernel<false><<<dim3(KVD / 128, S_LEN / 128), blk, 0, stream>>>(
      value, Wv, bv, v_raw, S_LEN, KVD, DIM_);
  // RoPE + pack to f16 head-major
  rope_pack_kernel<<<dim3(S_LEN), blk, 0, stream>>>(q_raw, k_raw, v_raw, q16,
                                                    k16, v16);
  // fused causal GQA flash attention
  attention_kernel<<<dim3(16, 16), blk, 0, stream>>>(q16, k16, v16, att16);
  // output projection (f16 A path), fp32 result straight to d_out
  gemm_bias_kernel<true><<<dim3(DIM_ / 128, S_LEN / 128), blk, 0, stream>>>(
      att16, Wo, bo, out, S_LEN, DIM_, DIM_);
}
